// DynamisCropClassifier_38242388803826
// MI455X (gfx1250) — matrix operand order repaired and checked by reading, not compiled, not run on hardware
//
// DynamisCropClassifier for MI455X (gfx1250, wave32, WMMA).
// f16 WMMA (f32 accum) for all large GEMMs + attention core; f32 VALU for the
// latency-bound Kalman/executor scan, LN, softmax, physics tails.
// v2: fragment-ordered V tile in LDS (b128 B-frag loads), bank-conflict-free
// padded softmax scratch, full-wave softmax via shfl_xor, direct f16 attention
// output (drops a 16MB buffer + convert kernel).
#include <hip/hip_runtime.h>
#include <cstdint>
#include <cstddef>

typedef __attribute__((ext_vector_type(16))) _Float16 v16h;
typedef __attribute__((ext_vector_type(8)))  float    v8f;

#define B_DIM 256
#define T_DIM 256
#define FIN   17
#define HID   64
#define S_DIM 7
#define HEADS 4
#define DH    16
#define PS_LD (T_DIM + 4)   // padded row stride: lane-to-lane bank stride 4

static __device__ __forceinline__ v8f wmma_f16(v16h a, v16h b, v8f c) {
  // v_wmma_f32_16x16x32_f16 (neg_a, A, neg_b, B, c_mod, C, reuse_a, reuse_b)
  return __builtin_amdgcn_wmma_f32_16x16x32_f16(false, a, false, b, (short)0, c, false, false);
}

// ---------------------------------------------------------------------------
// Weight repack: B[k][n] = W[n][k], fragment order per 32x16 K-tile so every
// B-fragment in the GEMM is two contiguous 16B loads per lane.
// ---------------------------------------------------------------------------
__global__ void pack_b_kernel(const float* __restrict__ W, _Float16* __restrict__ out,
                              int Kin, int N) {
  int ntiles = N >> 4;
  int tile = blockIdx.x;
  int kt = tile / ntiles, nt = tile % ntiles;
  int lane = threadIdx.x;                  // 32 threads/block
  int n = nt * 16 + (lane & 15);
  int kbase = kt * 32 + ((lane < 16) ? 0 : 16);
  _Float16* dst = out + (size_t)tile * 512 + lane * 16;
  for (int hh = 0; hh < 16; ++hh) {
    int kk = kbase + hh;
    float v = (kk < Kin) ? W[(size_t)n * Kin + kk] : 0.0f;
    dst[hh] = (_Float16)v;
  }
}

// ---------------------------------------------------------------------------
// Generic WMMA GEMM: C[M x N] = A16[M x Kpad] @ Bpack + bias.
// One wave per 16-row M-tile; K accumulated in registers; N looped in 16-tiles.
// ---------------------------------------------------------------------------
#define GEMM_WAVES 4
__global__ void gemm_wmma_kernel(const _Float16* __restrict__ A,
                                 const _Float16* __restrict__ Bp,
                                 const float* __restrict__ bias,
                                 float* __restrict__ C32,
                                 _Float16* __restrict__ C16,
                                 int M, int N, int Kpad) {
  int lane = threadIdx.x & 31;
  int wave = threadIdx.x >> 5;
  int mt = blockIdx.x * GEMM_WAVES + wave;
  if (mt * 16 >= M) return;
  int m0 = mt * 16;
  int ntiles = N >> 4, ktiles = Kpad >> 5;
  int colL = lane & 15;
  int rowA = m0 + colL;
  int kOff = (lane < 16) ? 0 : 16;              // A-layout K split per half-wave
  int rowBase = m0 + ((lane < 16) ? 0 : 8);     // C/D: hi lanes own rows +8
  for (int nt = 0; nt < ntiles; ++nt) {
    int n = nt * 16 + colL;
    float bn = bias ? bias[n] : 0.0f;
    v8f c; for (int i = 0; i < 8; ++i) c[i] = bn;
    for (int kt = 0; kt < ktiles; ++kt) {
      union { v16h v; float4 f[2]; } a, b;
      const _Float16* ap = A + (size_t)rowA * Kpad + kt * 32 + kOff;
      a.f[0] = *(const float4*)(ap);            // halves 0..7  (K: +0..7)
      a.f[1] = *(const float4*)(ap + 8);        // halves 8..15 (K: +8..15)
      const _Float16* bp = Bp + ((size_t)(kt * ntiles + nt) * 512) + lane * 16;
      b.f[0] = *(const float4*)(bp);
      b.f[1] = *(const float4*)(bp + 8);
      c = wmma_f16(a.v, b.v, c);
    }
    for (int r = 0; r < 8; ++r) {
      size_t idx = (size_t)(rowBase + r) * N + n;
      if (C32) C32[idx] = c[r];
      if (C16) C16[idx] = (_Float16)c[r];
    }
  }
}

// ---------------------------------------------------------------------------
// Input projection + LayerNorm: h = LN(x @ W_in^T + b_in). One row per block.
// ---------------------------------------------------------------------------
__global__ void input_proj_ln_kernel(const float* __restrict__ x,
                                     const float* __restrict__ W_in, const float* __restrict__ b_in,
                                     const float* __restrict__ g_in, const float* __restrict__ be_in,
                                     float* __restrict__ h) {
  int row = blockIdx.x;
  int tid = threadIdx.x;                        // 64 threads
  __shared__ float sx[FIN];
  __shared__ float red[HID];
  if (tid < FIN) sx[tid] = x[(size_t)row * FIN + tid];
  __syncthreads();
  float a = b_in[tid];
  const float* w = W_in + (size_t)tid * FIN;
  for (int j = 0; j < FIN; ++j) a += w[j] * sx[j];
  red[tid] = a; __syncthreads();
  for (int off = 32; off > 0; off >>= 1) { if (tid < off) red[tid] += red[tid + off]; __syncthreads(); }
  float mean = red[0] * (1.0f / HID);
  __syncthreads();
  float d = a - mean;
  red[tid] = d * d; __syncthreads();
  for (int off = 32; off > 0; off >>= 1) { if (tid < off) red[tid] += red[tid + off]; __syncthreads(); }
  float var = red[0] * (1.0f / HID);
  h[(size_t)row * HID + tid] = d * rsqrtf(var + 1e-5f) * g_in[tid] + be_in[tid];
}

// ---------------------------------------------------------------------------
// Sequential Kalman + executor scan. One 64-thread block per batch element.
// Wh (64x135, 34.5KB) staged in LDS once; 7x7 solve by thread 0 (Gauss-Jordan).
// ---------------------------------------------------------------------------
__global__ void scan_kernel(const float* __restrict__ h,
                            const float* __restrict__ Wh, const float* __restrict__ bhv,
                            const float* __restrict__ Wm, const float* __restrict__ bm,
                            const float* __restrict__ A,
                            float* __restrict__ straj, float* __restrict__ innov_out,
                            float* __restrict__ ptraj, float* __restrict__ xmkm,
                            float* __restrict__ unc) {
  int b = blockIdx.x, tid = threadIdx.x;        // 64 threads
  __shared__ float sWh[HID * 135];
  __shared__ float sWm[S_DIM * HID];
  __shared__ float sA[49], sP[49], sM1[49], sPp[49], sSinv[49], sK[49], sPn[49];
  __shared__ float sx[S_DIM], sxp[S_DIM], sinn[S_DIM], sbm[S_DIM];
  __shared__ float sHt[HID];
  __shared__ float sHex[2][HID];
  for (int i = tid; i < HID * 135; i += 64) sWh[i] = Wh[i];
  for (int i = tid; i < S_DIM * HID; i += 64) sWm[i] = Wm[i];
  if (tid < 49) { sA[tid] = A[tid]; sP[tid] = (tid % 8 == 0) ? 0.1f : 0.0f; }
  if (tid < S_DIM) { sx[tid] = 0.0f; sbm[tid] = bm[tid]; }
  sHex[0][tid] = 0.0f;
  const float bh_t = bhv[tid];
  int cur = 0;
  for (int t = 0; t < T_DIM; ++t) {
    __syncthreads();
    sHt[tid] = h[((size_t)(b * T_DIM + t)) * HID + tid];
    if (tid < S_DIM) {                          // x_pred = A @ x
      float v = 0.0f;
      for (int j = 0; j < 7; ++j) v += sA[tid * 7 + j] * sx[j];
      sxp[tid] = v;
    }
    if (tid < 49) {                             // M1 = A @ P
      int i = tid / 7, k = tid % 7;
      float m = 0.0f;
      for (int j = 0; j < 7; ++j) m += sA[i * 7 + j] * sP[j * 7 + k];
      sM1[tid] = m;
    }
    __syncthreads();
    if (tid < 49) {                             // P_pred = M1 @ A^T + Q (Q=0.05 I)
      int i = tid / 7, l = tid % 7;
      float v = 0.0f;
      for (int k = 0; k < 7; ++k) v += sM1[i * 7 + k] * sA[l * 7 + k];
      sPp[tid] = v + ((i == l) ? 0.05f : 0.0f);
    }
    // executor MLP: tanh(Wh @ [h_t, x_pred, h_exec] + bh)
    float acc = bh_t;
    const float* w = &sWh[tid * 135];
    for (int j = 0; j < HID; ++j) acc += w[j] * sHt[j];
    for (int j = 0; j < S_DIM; ++j) acc += w[64 + j] * sxp[j];
    const float* hp = sHex[cur];
    for (int j = 0; j < HID; ++j) acc += w[71 + j] * hp[j];
    float hn = tanhf(acc);
    sHex[cur ^ 1][tid] = hn;
    __syncthreads();
    if (tid < S_DIM) {                          // z = Wm @ h_exec + bm ; innov = z - x_pred
      float z = sbm[tid];
      const float* wm = &sWm[tid * HID];
      const float* he = sHex[cur ^ 1];
      for (int j = 0; j < HID; ++j) z += wm[j] * he[j];
      sinn[tid] = z - sxp[tid];
    }
    __syncthreads();
    if (tid == 0) {                             // Sinv = (P_pred + R)^-1, R = 0.05 I
      float Mx[7][14];
      for (int i = 0; i < 7; ++i)
        for (int j = 0; j < 7; ++j) {
          Mx[i][j] = sPp[i * 7 + j] + ((i == j) ? 0.05f : 0.0f);
          Mx[i][7 + j] = (i == j) ? 1.0f : 0.0f;
        }
      for (int c = 0; c < 7; ++c) {
        int piv = c; float bv = fabsf(Mx[c][c]);
        for (int r = c + 1; r < 7; ++r) { float av = fabsf(Mx[r][c]); if (av > bv) { bv = av; piv = r; } }
        if (piv != c) for (int j = 0; j < 14; ++j) { float t2 = Mx[c][j]; Mx[c][j] = Mx[piv][j]; Mx[piv][j] = t2; }
        float inv = 1.0f / Mx[c][c];
        for (int j = 0; j < 14; ++j) Mx[c][j] *= inv;
        for (int r = 0; r < 7; ++r) if (r != c) {
          float f = Mx[r][c];
          for (int j = 0; j < 14; ++j) Mx[r][j] -= f * Mx[c][j];
        }
      }
      for (int i = 0; i < 7; ++i)
        for (int j = 0; j < 7; ++j) sSinv[i * 7 + j] = Mx[i][7 + j];
    }
    __syncthreads();
    if (tid < 49) {                             // K = P_pred @ Sinv
      int i = tid / 7, j = tid % 7;
      float v = 0.0f;
      for (int k = 0; k < 7; ++k) v += sPp[i * 7 + k] * sSinv[k * 7 + j];
      sK[tid] = v;
    }
    __syncthreads();
    if (tid < S_DIM) {
      float xn = sxp[tid];
      for (int j = 0; j < 7; ++j) xn += sK[tid * 7 + j] * sinn[j];
      sx[tid] = xn;
      size_t o = ((size_t)(b * T_DIM + t)) * S_DIM + tid;
      straj[o] = xn;
      innov_out[o] = sinn[tid];
    }
    if (tid < 49) {                             // P_new = (I - K) @ P_pred
      int i = tid / 7, j = tid % 7;
      float v = sPp[tid];
      for (int k = 0; k < 7; ++k) v -= sK[i * 7 + k] * sPp[k * 7 + j];
      sPn[tid] = v;
    }
    __syncthreads();
    if (tid < 49) sP[tid] = sPn[tid];
    if (tid < S_DIM) ptraj[((size_t)(b * T_DIM + t)) * S_DIM + tid] = sPn[tid * 8];
    cur ^= 1;
  }
  __syncthreads();
  if (tid < S_DIM) xmkm[b * S_DIM + tid] = sx[tid];
  if (tid == 0) { float tr = 0.0f; for (int i = 0; i < 7; ++i) tr += sP[i * 8]; unc[b] = tr; }
}

// ---------------------------------------------------------------------------
// Build [h | state_traj | 0-pad] f16 rows (K padded 71 -> 96) for the Wp GEMM.
// ---------------------------------------------------------------------------
__global__ void build_hs16_kernel(const float* __restrict__ h, const float* __restrict__ straj,
                                  _Float16* __restrict__ hs16, int nrows) {
  int idx = blockIdx.x * blockDim.x + threadIdx.x;
  if (idx >= nrows * 96) return;
  int row = idx / 96, c = idx % 96;
  float v;
  if (c < 64)      v = h[(size_t)row * 64 + c];
  else if (c < 71) v = straj[(size_t)row * 7 + (c - 64)];
  else             v = 0.0f;
  hs16[idx] = (_Float16)v;
}

// ---------------------------------------------------------------------------
// Fused attention core per (b, head):
//  - logits via WMMA (dh=16 zero-padded to K=32), B-frag = 2x b128 from Ks
//  - masked scaled softmax in f32 LDS (padded stride, conflict-free), both
//    half-waves cooperate per row via shfl_xor(16)
//  - P@V via WMMA over K=256 with a fragment-ordered Vt tile (2x b128 B-frags)
//  - f16 output (direct operand for the Wo GEMM)
// ---------------------------------------------------------------------------
#define ATT_WAVES 2
__global__ void attn_core_kernel(const _Float16* __restrict__ q16,
                                 const _Float16* __restrict__ k16,
                                 const _Float16* __restrict__ v16,
                                 const unsigned char* __restrict__ mask,
                                 _Float16* __restrict__ o16) {
  __shared__ __align__(16) _Float16 Ks[T_DIM][DH];       // 8KB, row-major K tile
  __shared__ __align__(16) _Float16 Vs[T_DIM][DH];       // 8KB, staging
  __shared__ __align__(16) _Float16 Vt[8][32][16];       // 8KB, fragment order
  __shared__ float Ps[ATT_WAVES][16][PS_LD];             // padded: bank stride 4
  int bh = blockIdx.x; int b = bh >> 2; int hd = bh & 3;
  int tid = threadIdx.x; int lane = tid & 31; int wave = tid >> 5;
  for (int i = tid; i < T_DIM * 2; i += blockDim.x) {    // stage K,V head slices
    int row = i >> 1; int c8 = (i & 1) * 8;
    size_t g = ((size_t)(b * T_DIM + row)) * HID + hd * DH + c8;
    *(float4*)(&Ks[row][c8]) = *(const float4*)(k16 + g);
    *(float4*)(&Vs[row][c8]) = *(const float4*)(v16 + g);
  }
  __syncthreads();
  // Vt[kc][L][hh] = V[kc*32 + (L<16?0:16) + hh][L&15]  (B-frag layout for K=32)
  for (int i = tid; i < 8 * 32 * 2; i += blockDim.x) {
    int kc = i >> 6; int rem = i & 63; int L = rem >> 1; int hh0 = (rem & 1) * 8;
    int kb = kc * 32 + ((L < 16) ? 0 : 16);
    int d = L & 15;
    for (int j = 0; j < 8; ++j) Vt[kc][L][hh0 + j] = Vs[kb + hh0 + j][d];
  }
  __syncthreads();
  int colL = lane & 15;
  bool hi = lane >= 16;
  for (int qt = wave; qt < 16; qt += ATT_WAVES) {
    // A = Q tile (16 x 16, K-padded to 32): two 16B global loads, hi half zero.
    union { v16h v; float4 f[2]; } aq;
    {
      int qrow = qt * 16 + colL;
      const _Float16* qp = q16 + ((size_t)(b * T_DIM + qrow)) * HID + hd * DH;
      aq.f[0] = hi ? make_float4(0.f, 0.f, 0.f, 0.f) : *(const float4*)qp;
      aq.f[1] = hi ? make_float4(0.f, 0.f, 0.f, 0.f) : *(const float4*)(qp + 8);  // hi lanes: K 16..31 = pad
    }
    for (int kt = 0; kt < 16; ++kt) {                   // logits tiles over k-pos
      union { v16h v; float4 f[2]; } bk;
      if (!hi) {
        const _Float16* kp = &Ks[kt * 16 + colL][0];    // B[K=d][N=kpos]=Kmat[kpos][d]
        bk.f[0] = *(const float4*)kp;
        bk.f[1] = *(const float4*)(kp + 8);
      } else {
        bk.f[0] = make_float4(0.f, 0.f, 0.f, 0.f);      // K 16..31 = pad
        bk.f[1] = make_float4(0.f, 0.f, 0.f, 0.f);
      }
      v8f c; for (int i = 0; i < 8; ++i) c[i] = 0.0f;
      c = wmma_f16(aq.v, bk.v, c);
      int kpos = kt * 16 + colL;
      bool mok = mask[(size_t)b * T_DIM + kpos] != 0;
      int mrow = hi ? 8 : 0;
      for (int r = 0; r < 8; ++r)
        Ps[wave][mrow + r][kpos] = mok ? c[r] * 0.25f : -1e9f;  // 1/sqrt(16)
    }
    {   // softmax: lanes L and L+16 cooperate on row L (128 cols each)
      float* row = Ps[wave][colL];
      int base = hi ? 128 : 0;
      float mx = -3.0e38f;
      for (int j = 0; j < 128; ++j) mx = fmaxf(mx, row[base + j]);
      mx = fmaxf(mx, __shfl_xor(mx, 16, 32));
      float s = 0.0f;
      for (int j = 0; j < 128; ++j) { float e = expf(row[base + j] - mx); row[base + j] = e; s += e; }
      s += __shfl_xor(s, 16, 32);
      float inv = 1.0f / s;
      for (int j = 0; j < 128; ++j) row[base + j] *= inv;
    }
    v8f acc; for (int i = 0; i < 8; ++i) acc[i] = 0.0f; // O = P @ V, K = 256
    for (int kc = 0; kc < 8; ++kc) {
      union { v16h v; _Float16 hh[16]; } ap;
      const float* prow = Ps[wave][colL];
      int k0 = kc * 32 + (hi ? 16 : 0);
      for (int j = 0; j < 8; ++j) ap.hh[j]     = (_Float16)prow[k0 + j];
      for (int j = 0; j < 8; ++j) ap.hh[8 + j] = (_Float16)prow[k0 + 8 + j];
      union { v16h v; float4 f[2]; } bv;
      const _Float16* vp = &Vt[kc][lane][0];            // fragment order: 2x b128
      bv.f[0] = *(const float4*)vp;
      bv.f[1] = *(const float4*)(vp + 8);
      acc = wmma_f16(ap.v, bv.v, acc);
    }
    {
      int mrow = hi ? 8 : 0;
      for (int r = 0; r < 8; ++r) {
        int q = qt * 16 + mrow + r;
        o16[((size_t)(b * T_DIM + q)) * HID + hd * DH + colL] = (_Float16)acc[r];
      }
    }
  }
}

// ---------------------------------------------------------------------------
// chaos = clip(mean(innov^2),0,10)/10 ; gate = sigmoid(Wg@[chaos,hurst]+bg)
// ---------------------------------------------------------------------------
__global__ void chaos_gate_kernel(const float* __restrict__ innov, const float* __restrict__ hurst,
                                  const float* __restrict__ Wg, const float* __restrict__ bg,
                                  float* __restrict__ gate) {
  int b = blockIdx.x, tid = threadIdx.x;        // 256 threads
  __shared__ float red[256];
  float s = 0.0f;
  for (int i = tid; i < T_DIM * S_DIM; i += 256) {
    float v = innov[(size_t)b * T_DIM * S_DIM + i]; s += v * v;
  }
  red[tid] = s; __syncthreads();
  for (int off = 128; off > 0; off >>= 1) { if (tid < off) red[tid] += red[tid + off]; __syncthreads(); }
  float chaos = fminf(fmaxf(red[0] * (1.0f / (T_DIM * S_DIM)), 0.0f), 10.0f) * 0.1f;
  if (tid < HID) {
    float g = Wg[tid * 2] * chaos + Wg[tid * 2 + 1] * hurst[b] + bg[tid];
    gate[b * HID + tid] = 1.0f / (1.0f + expf(-g));
  }
}

__global__ void gate_apply_kernel(const float* __restrict__ o, const float* __restrict__ gate,
                                  float* __restrict__ ao) {
  int i = blockIdx.x * blockDim.x + threadIdx.x;        // B*T*HID
  int u = i & 63;
  int b = i >> 14;                                      // / (T_DIM*HID)
  ao[i] = o[i] * gate[b * HID + u];
}

// masked mean-pool over T then LayerNorm
__global__ void pool_ln_kernel(const float* __restrict__ ao, const unsigned char* __restrict__ mask,
                               const float* __restrict__ gp, const float* __restrict__ bep,
                               float* __restrict__ pooled) {
  int b = blockIdx.x, tid = threadIdx.x;        // 64 threads
  float s = 0.0f, cnt = 0.0f;
  for (int t = 0; t < T_DIM; ++t) {
    float m = mask[b * T_DIM + t] ? 1.0f : 0.0f;
    s += m * ao[((size_t)(b * T_DIM + t)) * HID + tid];
    cnt += m;
  }
  float p = s / fmaxf(cnt, 1e-6f);
  __shared__ float red[HID];
  red[tid] = p; __syncthreads();
  for (int off = 32; off > 0; off >>= 1) { if (tid < off) red[tid] += red[tid + off]; __syncthreads(); }
  float mean = red[0] * (1.0f / HID);
  __syncthreads();
  float d = p - mean;
  red[tid] = d * d; __syncthreads();
  for (int off = 32; off > 0; off >>= 1) { if (tid < off) red[tid] += red[tid + off]; __syncthreads(); }
  float var = red[0] * (1.0f / HID);
  pooled[b * HID + tid] = d * rsqrtf(var + 1e-5f) * gp[tid] + bep[tid];
}

// physics feature vector + its LayerNorm
__global__ void phys_kernel(const float* __restrict__ innov, const float* __restrict__ ptraj,
                            const float* __restrict__ xmkm, const float* __restrict__ hurst,
                            const unsigned char* __restrict__ mask,
                            const float* __restrict__ gph, const float* __restrict__ beph,
                            float* __restrict__ phys_out, float* __restrict__ physn) {
  int b = blockIdx.x, t = threadIdx.x;          // 256 threads
  __shared__ float rs[256];
  __shared__ int   ri[256];
  __shared__ float sv[8];
  float mf = mask[b * T_DIM + t] ? 1.0f : 0.0f;
  float im = 0.0f, trp = 0.0f;
  for (int s = 0; s < S_DIM; ++s) {
    float v = innov[((size_t)(b * T_DIM + t)) * S_DIM + s]; im += v * v;
    trp += ptraj[((size_t)(b * T_DIM + t)) * S_DIM + s];
  }
  rs[t] = mf; __syncthreads();
  for (int off = 128; off > 0; off >>= 1) { if (t < off) rs[t] += rs[t + off]; __syncthreads(); }
  if (t == 0) sv[0] = fmaxf(rs[0], 1.0f);
  __syncthreads();
  float mfs = sv[0];
  rs[t] = im * mf; __syncthreads();
  for (int off = 128; off > 0; off >>= 1) { if (t < off) rs[t] += rs[t + off]; __syncthreads(); }
  if (t == 0) sv[1] = rs[0] / mfs;
  __syncthreads();
  float imean = sv[1];
  rs[t] = (mf > 0.0f) ? im : -3.0e38f; ri[t] = t; __syncthreads();
  for (int off = 128; off > 0; off >>= 1) {
    if (t < off) {
      if (rs[t + off] > rs[t]) { rs[t] = rs[t + off]; ri[t] = ri[t + off]; }
      else if (rs[t + off] == rs[t] && ri[t + off] < ri[t]) ri[t] = ri[t + off];
    }
    __syncthreads();
  }
  if (t == 0) { sv[2] = rs[0]; sv[3] = (float)ri[0] * (1.0f / (T_DIM - 1)); }
  __syncthreads();
  float cen = (im - imean) * mf;
  rs[t] = cen * cen; __syncthreads();
  for (int off = 128; off > 0; off >>= 1) { if (t < off) rs[t] += rs[t + off]; __syncthreads(); }
  if (t == 0) sv[4] = sqrtf(fmaxf(rs[0] / mfs, 1e-8f));
  __syncthreads();
  rs[t] = trp; __syncthreads();
  for (int off = 128; off > 0; off >>= 1) { if (t < off) rs[t] += rs[t + off]; __syncthreads(); }
  if (t == 0) sv[5] = rs[0] * (1.0f / T_DIM);
  __syncthreads();
  rs[t] = trp; __syncthreads();
  for (int off = 128; off > 0; off >>= 1) { if (t < off) rs[t] = fmaxf(rs[t], rs[t + off]); __syncthreads(); }
  if (t == 0) sv[6] = rs[0];
  __syncthreads();
  rs[t] = trp; __syncthreads();
  for (int off = 128; off > 0; off >>= 1) { if (t < off) rs[t] = fminf(rs[t], rs[t + off]); __syncthreads(); }
  if (t == 0) sv[7] = rs[0];
  __syncthreads();
  if (t == 0) {
    float ph[15];
    ph[0] = imean; ph[1] = sv[2]; ph[2] = sv[4]; ph[3] = sv[3];
    for (int s = 0; s < 7; ++s) ph[4 + s] = xmkm[b * 7 + s];
    ph[11] = sv[5]; ph[12] = sv[6]; ph[13] = sv[6] - sv[7]; ph[14] = hurst[b];
    float m = 0.0f; for (int i = 0; i < 15; ++i) m += ph[i];
    m *= (1.0f / 15.0f);
    float v = 0.0f; for (int i = 0; i < 15; ++i) { float d = ph[i] - m; v += d * d; }
    v *= (1.0f / 15.0f);
    float inv = rsqrtf(v + 1e-5f);
    for (int i = 0; i < 15; ++i) {
      phys_out[b * 15 + i] = ph[i];
      physn[b * 15 + i] = (ph[i] - m) * inv * gph[i] + beph[i];
    }
  }
}

__global__ void hcat_kernel(const float* __restrict__ pooled, const float* __restrict__ physn,
                            _Float16* __restrict__ hcat16) {
  int idx = blockIdx.x * blockDim.x + threadIdx.x;      // B_DIM*96
  if (idx >= B_DIM * 96) return;
  int b = idx / 96, c = idx % 96;
  float v = (c < 64) ? pooled[b * 64 + c] : ((c < 79) ? physn[b * 15 + (c - 64)] : 0.0f);
  hcat16[idx] = (_Float16)v;
}

__global__ void head_kernel(const float* __restrict__ h1, const float* __restrict__ W2,
                            const float* __restrict__ b2, float* __restrict__ crop) {
  int b = blockIdx.x, tid = threadIdx.x;        // 64 threads
  __shared__ float sg[HID];
  float v = h1[b * HID + tid];
  float g = 0.5f * v * (1.0f + erff(v * 0.70710678118654752f));   // exact gelu
  sg[tid] = g; __syncthreads();
  if (tid < 3) {
    float a = b2[tid];
    for (int j = 0; j < HID; ++j) a += W2[tid * HID + j] * sg[j];
    crop[b * 3 + tid] = a;
  }
}

__global__ void pheno_kernel(const float* __restrict__ ao, const float* __restrict__ Wph,
                             const float* __restrict__ bph, float* __restrict__ out) {
  int idx = blockIdx.x * blockDim.x + threadIdx.x;      // B*T*S
  if (idx >= B_DIM * T_DIM * S_DIM) return;
  int row = idx / S_DIM, s = idx % S_DIM;
  float a = bph[s];
  const float* ar = ao + (size_t)row * HID;
  const float* w = Wph + s * HID;
  for (int j = 0; j < HID; ++j) a += w[j] * ar[j];
  out[idx] = a;
}

// ---------------------------------------------------------------------------
extern "C" void kernel_launch(void* const* d_in, const int* in_sizes, int n_in,
                              void* d_out, int out_size, void* d_ws, size_t ws_size,
                              hipStream_t stream) {
  (void)in_sizes; (void)n_in; (void)out_size; (void)ws_size;
  const float* x      = (const float*)d_in[0];
  const unsigned char* mask = (const unsigned char*)d_in[1];   // jnp bool -> 1 byte
  const float* hurst  = (const float*)d_in[2];
  const float* W_in   = (const float*)d_in[3];
  const float* b_in   = (const float*)d_in[4];
  const float* g_in   = (const float*)d_in[5];
  const float* be_in  = (const float*)d_in[6];
  const float* Wh     = (const float*)d_in[7];
  const float* bh     = (const float*)d_in[8];
  const float* Wm     = (const float*)d_in[9];
  const float* bm     = (const float*)d_in[10];
  const float* A      = (const float*)d_in[11];
  const float* Wp     = (const float*)d_in[12];
  const float* bp     = (const float*)d_in[13];
  const float* Wq     = (const float*)d_in[14];
  const float* bq     = (const float*)d_in[15];
  const float* Wk     = (const float*)d_in[16];
  const float* bk     = (const float*)d_in[17];
  const float* Wv     = (const float*)d_in[18];
  const float* bv     = (const float*)d_in[19];
  const float* Wo     = (const float*)d_in[20];
  const float* bo     = (const float*)d_in[21];
  const float* Wg     = (const float*)d_in[22];
  const float* bg     = (const float*)d_in[23];
  const float* g_pool = (const float*)d_in[24];
  const float* be_pool= (const float*)d_in[25];
  const float* g_phys = (const float*)d_in[26];
  const float* be_phys= (const float*)d_in[27];
  const float* W1     = (const float*)d_in[28];
  const float* b1     = (const float*)d_in[29];
  const float* W2     = (const float*)d_in[30];
  const float* b2     = (const float*)d_in[31];
  const float* Wph    = (const float*)d_in[32];
  const float* bph    = (const float*)d_in[33];

  // d_out layout (tuple return order, f32):
  float* out   = (float*)d_out;
  float* crop  = out;                     // 256*3
  float* pheno = out + 768;               // 256*256*7
  float* innov = out + 459520;            // 256*256*7
  float* unc   = out + 918272;            // 256
  float* xmkm  = out + 918528;            // 256*7
  float* straj = out + 920320;            // 256*256*7
  float* ptraj = out + 1379072;           // 256*256*7
  float* phys  = out + 1837824;           // 256*15

  char* ws = (char*)d_ws;
  const size_t OFF_WPK = 0,        OFF_WQK = 12288, OFF_WKK = 20480, OFF_WVK = 28672,
               OFF_WOK = 36864,    OFF_W1K = 45056;
  const size_t OFF_H    = 65536;
  const size_t OFF_HS16 = OFF_H    + 16777216;
  const size_t OFF_AI16 = OFF_HS16 + 12582912;
  const size_t OFF_Q16  = OFF_AI16 + 8388608;
  const size_t OFF_K16  = OFF_Q16  + 8388608;
  const size_t OFF_V16  = OFF_K16  + 8388608;
  const size_t OFF_O16  = OFF_V16  + 8388608;   // attention output (f16)
  const size_t OFF_OP32 = OFF_O16  + 8388608;   // after Wo (f32)
  const size_t OFF_AO32 = OFF_OP32 + 16777216;  // gated attn_out (f32)
  const size_t OFF_GATE = OFF_AO32 + 16777216;
  const size_t OFF_POOL = OFF_GATE + 65536;
  const size_t OFF_PHN  = OFF_POOL + 65536;
  const size_t OFF_HCAT = OFF_PHN  + 15360;
  const size_t OFF_H1   = OFF_HCAT + 49152;     // total ~100MB

  _Float16* WpP  = (_Float16*)(ws + OFF_WPK);
  _Float16* WqP  = (_Float16*)(ws + OFF_WQK);
  _Float16* WkP  = (_Float16*)(ws + OFF_WKK);
  _Float16* WvP  = (_Float16*)(ws + OFF_WVK);
  _Float16* WoP  = (_Float16*)(ws + OFF_WOK);
  _Float16* W1P  = (_Float16*)(ws + OFF_W1K);
  float*    hbuf = (float*)(ws + OFF_H);
  _Float16* hs16 = (_Float16*)(ws + OFF_HS16);
  _Float16* ai16 = (_Float16*)(ws + OFF_AI16);
  _Float16* q16  = (_Float16*)(ws + OFF_Q16);
  _Float16* k16  = (_Float16*)(ws + OFF_K16);
  _Float16* v16  = (_Float16*)(ws + OFF_V16);
  _Float16* o16  = (_Float16*)(ws + OFF_O16);
  float*    op32 = (float*)(ws + OFF_OP32);
  float*    ao32 = (float*)(ws + OFF_AO32);
  float*    gate = (float*)(ws + OFF_GATE);
  float*    pool = (float*)(ws + OFF_POOL);
  float*    phn  = (float*)(ws + OFF_PHN);
  _Float16* hcat = (_Float16*)(ws + OFF_HCAT);
  float*    h1   = (float*)(ws + OFF_H1);

  const int NROWS = B_DIM * T_DIM;                  // 65536

  // 1) repack WMMA weight operands (fragment order)
  pack_b_kernel<<<dim3(3 * 4), dim3(32), 0, stream>>>(Wp, WpP, 71, 64);   // Kpad 96
  pack_b_kernel<<<dim3(2 * 4), dim3(32), 0, stream>>>(Wq, WqP, 64, 64);
  pack_b_kernel<<<dim3(2 * 4), dim3(32), 0, stream>>>(Wk, WkP, 64, 64);
  pack_b_kernel<<<dim3(2 * 4), dim3(32), 0, stream>>>(Wv, WvP, 64, 64);
  pack_b_kernel<<<dim3(2 * 4), dim3(32), 0, stream>>>(Wo, WoP, 64, 64);
  pack_b_kernel<<<dim3(3 * 4), dim3(32), 0, stream>>>(W1, W1P, 79, 64);   // Kpad 96

  // 2) input projection + LN
  input_proj_ln_kernel<<<dim3(NROWS), dim3(64), 0, stream>>>(x, W_in, b_in, g_in, be_in, hbuf);

  // 3) sequential Kalman/executor scan (writes straj/innov/ptraj/xmkm/unc)
  scan_kernel<<<dim3(B_DIM), dim3(64), 0, stream>>>(hbuf, Wh, bh, Wm, bm, A,
                                                    straj, innov, ptraj, xmkm, unc);

  // 4) ai = [h | state] @ Wp^T + bp  (WMMA, f16 out)
  build_hs16_kernel<<<dim3(NROWS * 96 / 256), dim3(256), 0, stream>>>(hbuf, straj, hs16, NROWS);
  gemm_wmma_kernel<<<dim3(NROWS / 64), dim3(128), 0, stream>>>(hs16, WpP, bp, nullptr, ai16,
                                                               NROWS, 64, 96);
  // 5) Q,K,V projections (WMMA, f16 out)
  gemm_wmma_kernel<<<dim3(NROWS / 64), dim3(128), 0, stream>>>(ai16, WqP, bq, nullptr, q16, NROWS, 64, 64);
  gemm_wmma_kernel<<<dim3(NROWS / 64), dim3(128), 0, stream>>>(ai16, WkP, bk, nullptr, k16, NROWS, 64, 64);
  gemm_wmma_kernel<<<dim3(NROWS / 64), dim3(128), 0, stream>>>(ai16, WvP, bv, nullptr, v16, NROWS, 64, 64);

  // 6) fused attention core (WMMA logits + softmax + WMMA P@V), f16 out
  attn_core_kernel<<<dim3(B_DIM * HEADS), dim3(32 * ATT_WAVES), 0, stream>>>(q16, k16, v16, mask, o16);

  // 7) output projection Wo (WMMA, f32 out)
  gemm_wmma_kernel<<<dim3(NROWS / 64), dim3(128), 0, stream>>>(o16, WoP, bo, op32, nullptr, NROWS, 64, 64);

  // 8) physics gate, pooling, physics features
  chaos_gate_kernel<<<dim3(B_DIM), dim3(256), 0, stream>>>(innov, hurst, Wg, bg, gate);
  gate_apply_kernel<<<dim3(NROWS * HID / 256), dim3(256), 0, stream>>>(op32, gate, ao32);
  pool_ln_kernel<<<dim3(B_DIM), dim3(64), 0, stream>>>(ao32, mask, g_pool, be_pool, pool);
  phys_kernel<<<dim3(B_DIM), dim3(256), 0, stream>>>(innov, ptraj, xmkm, hurst, mask,
                                                     g_phys, be_phys, phys, phn);

  // 9) classifier head: gelu(hcat @ W1^T + b1) @ W2^T + b2
  hcat_kernel<<<dim3(B_DIM * 96 / 256), dim3(256), 0, stream>>>(pool, phn, hcat);
  gemm_wmma_kernel<<<dim3(B_DIM / 64), dim3(128), 0, stream>>>(hcat, W1P, b1, h1, nullptr, B_DIM, 64, 96);
  head_kernel<<<dim3(B_DIM), dim3(64), 0, stream>>>(h1, W2, b2, crop);

  // 10) phenophase head
  pheno_kernel<<<dim3(B_DIM * T_DIM * S_DIM / 256, 1, 1), dim3(256), 0, stream>>>(ao32, Wph, bph, pheno);
}
